// Attention_88201448391199
// MI455X (gfx1250) — matrix operand order repaired and checked by reading, not compiled, not run on hardware
//
#include <hip/hip_runtime.h>
#include <hip/hip_bf16.h>

typedef __attribute__((ext_vector_type(16))) _Float16 v16h;
typedef __attribute__((ext_vector_type(4)))  _Float16 v4h;
typedef __attribute__((ext_vector_type(8)))  float    v8f;
typedef __attribute__((ext_vector_type(4)))  unsigned int u32x4;
typedef __attribute__((ext_vector_type(2)))  unsigned int u32x2;
typedef __attribute__((ext_vector_type(8)))  int i32x8;
typedef __attribute__((ext_vector_type(4)))  int i32x4;

#define LDS_STRIDE 40  // halves: 32 data + 8 pad (TDM pad: 16B per 64B row)

// Toolchain-dependent TDM builtin arity (ROCm7.2/clang-22: 5 args; clang-23: 6)
#if defined(__clang_major__) && (__clang_major__ >= 23)
#define TDM_LOAD(g0, g1, g2, g3) \
  __builtin_amdgcn_tensor_load_to_lds((g0), (g1), (g2), (g3), (i32x8)0, 0)
#else
#define TDM_LOAD(g0, g1, g2, g3) \
  __builtin_amdgcn_tensor_load_to_lds((g0), (g1), (g2), (g3), 0)
#endif

// ---- TDM: DMA a [tile_d1 x tile_d0] f16 tile (row stride = stride_elems)
// from global into LDS at lds_byte_off, padded 16B per 64B row (-> stride 40h).
__device__ __forceinline__ void tdm_load_tile_f16(
    unsigned lds_byte_off, const _Float16* gsrc,
    unsigned tile_d0, unsigned tile_d1, unsigned long long stride_elems) {
  unsigned long long ga = (unsigned long long)(const void*)gsrc;
  u32x4 g0;
  g0[0] = 1u;                                                 // count=1, user D#
  g0[1] = lds_byte_off;                                       // lds_addr (bytes)
  g0[2] = (unsigned)ga;                                       // global_addr[31:0]
  g0[3] = (unsigned)((ga >> 32) & 0x01ffffffu) | (2u << 30);  // addr[56:32], type=2
  i32x8 g1;
  g1[0] = (int)((1u << 16)      // data_size = 2 bytes
              | (1u << 20)      // pad_enable
              | (3u << 22)      // pad_interval: 16 DWORDs = 64B
              | (3u << 25));    // pad_amount: 4 DWORDs = 16B
  g1[1] = (int)((tile_d0 & 0xffffu) << 16);                   // tensor_dim0[15:0]
  g1[2] = (int)((tile_d1 & 0xffffu) << 16);                   // tensor_dim1 lo
  g1[3] = (int)((tile_d0 & 0xffffu) << 16);                   // tile_dim0
  g1[4] = (int)(tile_d1 & 0xffffu);                           // tile_dim1 | tile_dim2=0
  g1[5] = (int)(unsigned)(stride_elems & 0xffffffffu);        // dim0_stride lo
  g1[6] = (int)(unsigned)((stride_elems >> 32) & 0xffffu);    // dim0_stride hi
  g1[7] = 0;
  i32x4 gz = (i32x4)0;
  TDM_LOAD(g0, g1, gz, gz);
}

// ---- WMMA fragment loaders (ISA 7.12.2, wave32, 16-bit operands) -----------
__device__ __forceinline__ v16h frag_a(const _Float16* row, int half8) {
  v16h f;
#pragma unroll
  for (int v = 0; v < 8; ++v) {
    int kb = (half8 << 1) + (v << 1);
    f[2 * v] = row[kb];
    f[2 * v + 1] = row[kb + 1];
  }
  return f;
}
__device__ __forceinline__ v16h frag_b(const _Float16* row, int half16) {
  v16h f;
#pragma unroll
  for (int v = 0; v < 8; ++v) {
    int kb = half16 + (v << 1);
    f[2 * v] = row[kb];
    f[2 * v + 1] = row[kb + 1];
  }
  return f;
}

// LDS double-buffer byte offsets (each tile = 64 rows * 40 halves = 5120 B)
#define AS_OFF0 0u
#define AS_OFF1 5120u
#define BS_OFF0 10240u
#define BS_OFF1 15360u
#define SMEM_BYTES 20480u

// ---- GEMM-NT: C[M,N] = scale * A[M,K] x B[N,K]^T, both f16 row-major -------
// TDM-fed, double-buffered (K-loop unrolled x2 so LDS offsets are immediates).
// MASKED: -inf where !(qm[row]&km[col]) (logit, C re-read -> RT stores).
// !MASKED: C is a final output -> NT stores; optional f16 mirror Ch (re-read).
template <bool MASKED>
__global__ __launch_bounds__(128) void gemm_nt_tdm(
    const _Float16* __restrict__ A, const _Float16* __restrict__ Bm,
    float* __restrict__ C, _Float16* __restrict__ Ch,
    int M, int N, int K, float scale,
    const unsigned char* __restrict__ qmask, const unsigned char* __restrict__ kmask) {
  extern __shared__ _Float16 smem[];
  const int tid = threadIdx.x;
  const int lane = tid & 31, wave = tid >> 5;
  const int wrow = (wave >> 1) << 5, wcol = (wave & 1) << 5;
  const int l15 = lane & 15, half8 = (lane >> 4) << 3, half16 = half8 << 1;

  const int ntiles = N >> 6;
  const int m0 = (blockIdx.x / ntiles) << 6;
  const int n0 = (blockIdx.x % ntiles) << 6;
  const int b = blockIdx.y;
  A += (size_t)b * M * K;
  Bm += (size_t)b * N * K;
  C += (size_t)b * M * N;
  const _Float16* Abase = A + (size_t)m0 * K;
  const _Float16* Bbase = Bm + (size_t)n0 * K;

  if (wave == 0) {  // TDM issue is per-wave; EXEC ignored
    tdm_load_tile_f16(AS_OFF0, Abase, 32u, 64u, (unsigned long long)K);
    tdm_load_tile_f16(BS_OFF0, Bbase, 32u, 64u, (unsigned long long)K);
    __builtin_amdgcn_s_wait_tensorcnt(0);
  }
  __syncthreads();

  v8f acc[2][2] = {};

  auto kstep = [&](int kk, bool more, unsigned aCur, unsigned bCur,
                   unsigned aNxt, unsigned bNxt) {
    if (wave == 0 && more) {  // DMA next K-slab while computing this one
      tdm_load_tile_f16(aNxt, Abase + kk + 32, 32u, 64u, (unsigned long long)K);
      tdm_load_tile_f16(bNxt, Bbase + kk + 32, 32u, 64u, (unsigned long long)K);
    }
    const _Float16* As = smem + (aCur >> 1);
    const _Float16* Bs = smem + (bCur >> 1);
    v16h a0 = frag_a(&As[(wrow + l15) * LDS_STRIDE], half8);
    v16h a1 = frag_a(&As[(wrow + 16 + l15) * LDS_STRIDE], half8);
    v16h b0 = frag_b(&Bs[(wcol + l15) * LDS_STRIDE], half16);
    v16h b1 = frag_b(&Bs[(wcol + 16 + l15) * LDS_STRIDE], half16);
    acc[0][0] = __builtin_amdgcn_wmma_f32_16x16x32_f16(false, a0, false, b0, (short)0, acc[0][0], false, false);
    acc[0][1] = __builtin_amdgcn_wmma_f32_16x16x32_f16(false, a0, false, b1, (short)0, acc[0][1], false, false);
    acc[1][0] = __builtin_amdgcn_wmma_f32_16x16x32_f16(false, a1, false, b0, (short)0, acc[1][0], false, false);
    acc[1][1] = __builtin_amdgcn_wmma_f32_16x16x32_f16(false, a1, false, b1, (short)0, acc[1][1], false, false);
    if (wave == 0 && more) __builtin_amdgcn_s_wait_tensorcnt(0);
    __syncthreads();
  };

  for (int kk = 0; kk < K; kk += 64) {  // K % 64 == 0 for all call sites
    kstep(kk, true, AS_OFF0, BS_OFF0, AS_OFF1, BS_OFF1);
    kstep(kk + 32, kk + 64 < K, AS_OFF1, BS_OFF1, AS_OFF0, BS_OFF0);
  }

#pragma unroll
  for (int i = 0; i < 2; ++i) {
#pragma unroll
    for (int j = 0; j < 2; ++j) {
      const int col = n0 + wcol + (j << 4) + l15;
#pragma unroll
      for (int r = 0; r < 8; ++r) {
        const int row = m0 + wrow + (i << 4) + r + half8;
        const size_t idx = (size_t)row * N + col;
        if constexpr (MASKED) {
          float val = acc[i][j][r] * scale;
          const unsigned char* qm = qmask + (size_t)b * M;
          const unsigned char* km = kmask + (size_t)b * N;
          if (!(qm[row] && km[col])) val = -__builtin_inff();
          C[idx] = val;  // logit: re-read by softmaxes -> regular temporal
        } else {
          float val = acc[i][j][r];
          __builtin_nontemporal_store(val, &C[idx]);  // final output
          if (Ch) Ch[(size_t)b * M * N + idx] = (_Float16)val;  // re-read -> RT
        }
      }
    }
  }
}

// ---- GEMM-NN: C[M,N] = A[M,K] x B[K,N], f16 operands. A via TDM; B tile
// transposed into LDS manually (batched b64 loads -> b16 scatter). N=128 here.
__global__ __launch_bounds__(128) void gemm_nn_tdm(
    const _Float16* __restrict__ A, const _Float16* __restrict__ Bm,
    float* __restrict__ C, int M, int N, int K) {
  extern __shared__ _Float16 smem[];
  const int tid = threadIdx.x;
  const int lane = tid & 31, wave = tid >> 5;
  const int wrow = (wave >> 1) << 5, wcol = (wave & 1) << 5;
  const int l15 = lane & 15, half8 = (lane >> 4) << 3, half16 = half8 << 1;

  const int ntiles = N >> 6;
  const int m0 = (blockIdx.x / ntiles) << 6;
  const int n0 = (blockIdx.x % ntiles) << 6;
  const int b = blockIdx.y;
  A += (size_t)b * M * K;
  Bm += (size_t)b * K * N;
  C += (size_t)b * M * N;
  const _Float16* Abase = A + (size_t)m0 * K;

  // Prologue: TDM A tile0; manual B tile0 (batched regs -> scatter).
  if (wave == 0)
    tdm_load_tile_f16(AS_OFF0, Abase, 32u, 64u, (unsigned long long)K);
  {
    u32x2 breg[4];
#pragma unroll
    for (int e = 0; e < 4; ++e) {
      int u = (e << 7) + tid;  // 512 u32x2's = 64n x 32k halves
      int c = u >> 4, nq = (u & 15) << 2;
      breg[e] = *(const u32x2*)(Bm + (size_t)c * N + n0 + nq);
    }
    _Float16* dst = smem + (BS_OFF0 >> 1);
#pragma unroll
    for (int e = 0; e < 4; ++e) {
      int u = (e << 7) + tid;
      int c = u >> 4, nq = (u & 15) << 2;
      const _Float16* h = (const _Float16*)&breg[e];
#pragma unroll
      for (int i = 0; i < 4; ++i) dst[(nq + i) * LDS_STRIDE + c] = h[i];
    }
  }
  if (wave == 0) __builtin_amdgcn_s_wait_tensorcnt(0);
  __syncthreads();

  v8f acc[2][2] = {};

  auto kstep = [&](int kk, bool more, unsigned aCur, unsigned bCur,
                   unsigned aNxt, unsigned bNxt) {
    u32x2 breg[4];
    if (more) {
      if (wave == 0)
        tdm_load_tile_f16(aNxt, Abase + kk + 32, 32u, 64u, (unsigned long long)K);
#pragma unroll
      for (int e = 0; e < 4; ++e) {  // batch global loads for next B tile
        int u = (e << 7) + tid;
        int c = u >> 4, nq = (u & 15) << 2;
        breg[e] = *(const u32x2*)(Bm + (size_t)(kk + 32 + c) * N + n0 + nq);
      }
    }
    const _Float16* As = smem + (aCur >> 1);
    const _Float16* Bs = smem + (bCur >> 1);
    v16h a0 = frag_a(&As[(wrow + l15) * LDS_STRIDE], half8);
    v16h a1 = frag_a(&As[(wrow + 16 + l15) * LDS_STRIDE], half8);
    v16h b0 = frag_b(&Bs[(wcol + l15) * LDS_STRIDE], half16);
    v16h b1 = frag_b(&Bs[(wcol + 16 + l15) * LDS_STRIDE], half16);
    acc[0][0] = __builtin_amdgcn_wmma_f32_16x16x32_f16(false, a0, false, b0, (short)0, acc[0][0], false, false);
    acc[0][1] = __builtin_amdgcn_wmma_f32_16x16x32_f16(false, a0, false, b1, (short)0, acc[0][1], false, false);
    acc[1][0] = __builtin_amdgcn_wmma_f32_16x16x32_f16(false, a1, false, b0, (short)0, acc[1][0], false, false);
    acc[1][1] = __builtin_amdgcn_wmma_f32_16x16x32_f16(false, a1, false, b1, (short)0, acc[1][1], false, false);
    if (more) {
      _Float16* dst = smem + (bNxt >> 1);
#pragma unroll
      for (int e = 0; e < 4; ++e) {
        int u = (e << 7) + tid;
        int c = u >> 4, nq = (u & 15) << 2;
        const _Float16* h = (const _Float16*)&breg[e];
#pragma unroll
        for (int i = 0; i < 4; ++i) dst[(nq + i) * LDS_STRIDE + c] = h[i];
      }
      if (wave == 0) __builtin_amdgcn_s_wait_tensorcnt(0);
    }
    __syncthreads();
  };

  for (int kk = 0; kk < K; kk += 64) {  // K % 64 == 0
    kstep(kk, true, AS_OFF0, BS_OFF0, AS_OFF1, BS_OFF1);
    kstep(kk + 32, kk + 64 < K, AS_OFF1, BS_OFF1, AS_OFF0, BS_OFF0);
  }

#pragma unroll
  for (int i = 0; i < 2; ++i) {
#pragma unroll
    for (int j = 0; j < 2; ++j) {
      const int col = n0 + wcol + (j << 4) + l15;
#pragma unroll
      for (int r = 0; r < 8; ++r) {
        const int row = m0 + wrow + (i << 4) + r + half8;
        __builtin_nontemporal_store(acc[i][j][r], &C[(size_t)row * N + col]);
      }
    }
  }
}

// ---- f32 -> f16 pack (vectorized) ------------------------------------------
__global__ __launch_bounds__(256) void cvt_f16(const float* __restrict__ src,
                                               _Float16* __restrict__ dst, int n4) {
  int i = blockIdx.x * 256 + threadIdx.x;
  if (i < n4) {
    float4 v = ((const float4*)src)[i];
    v4h h;
    h[0] = (_Float16)v.x; h[1] = (_Float16)v.y;
    h[2] = (_Float16)v.z; h[3] = (_Float16)v.w;
    ((v4h*)dst)[i] = h;
  }
}

// ---- Softmax over keys: f32 out (NT, final) + f16 mirror (RT, re-read) -----
__global__ __launch_bounds__(256) void softmax_rows(
    const float* __restrict__ logit, float* __restrict__ out,
    _Float16* __restrict__ out16, int L) {
  const size_t row = blockIdx.x;
  const float* x = logit + row * (size_t)L;
  float* y = out + row * (size_t)L;
  _Float16* y16 = out16 + row * (size_t)L;
  const int t = threadIdx.x;

  float v[8];
  float mx = -__builtin_inff();
#pragma unroll
  for (int i = 0; i < 8; ++i) { v[i] = x[t + (i << 8)]; mx = fmaxf(mx, v[i]); }

  __shared__ float red[8];
#pragma unroll
  for (int o = 16; o > 0; o >>= 1) mx = fmaxf(mx, __shfl_down(mx, o, 32));
  if ((t & 31) == 0) red[t >> 5] = mx;
  __syncthreads();
  mx = red[0];
#pragma unroll
  for (int w = 1; w < 8; ++w) mx = fmaxf(mx, red[w]);

  if (mx == -__builtin_inff()) {  // fully masked row -> zeros
#pragma unroll
    for (int i = 0; i < 8; ++i) {
      __builtin_nontemporal_store(0.f, &y[t + (i << 8)]);
      y16[t + (i << 8)] = (_Float16)0.f;
    }
    return;
  }
  float s = 0.f;
#pragma unroll
  for (int i = 0; i < 8; ++i) { v[i] = __expf(v[i] - mx); s += v[i]; }
  __syncthreads();
#pragma unroll
  for (int o = 16; o > 0; o >>= 1) s += __shfl_down(s, o, 32);
  if ((t & 31) == 0) red[t >> 5] = s;
  __syncthreads();
  s = 0.f;
#pragma unroll
  for (int w = 0; w < 8; ++w) s += red[w];
  const float inv = 1.f / s;
#pragma unroll
  for (int i = 0; i < 8; ++i) {
    float o = v[i] * inv;
    __builtin_nontemporal_store(o, &y[t + (i << 8)]);
    y16[t + (i << 8)] = (_Float16)o;
  }
}

// ---- Softmax over queries (columns) -> f16 kq_weight (re-read -> RT) -------
__global__ __launch_bounds__(256) void softmax_cols(
    const float* __restrict__ logit, _Float16* __restrict__ kq, int L) {
  const int b = blockIdx.y;
  const int k = (blockIdx.x << 5) + threadIdx.x;
  const int qy = threadIdx.y;
  const float* base = logit + (size_t)b * L * L;
  _Float16* outb = kq + (size_t)b * L * L;

  __shared__ float red[8][32];
  float mx = -__builtin_inff();
  for (int q = qy; q < L; q += 8) mx = fmaxf(mx, base[(size_t)q * L + k]);
  red[qy][threadIdx.x] = mx;
  __syncthreads();
#pragma unroll
  for (int w = 0; w < 8; ++w) mx = fmaxf(mx, red[w][threadIdx.x]);
  const bool dead = (mx == -__builtin_inff());
  __syncthreads();

  float s = 0.f;
  if (!dead)
    for (int q = qy; q < L; q += 8) s += __expf(base[(size_t)q * L + k] - mx);
  red[qy][threadIdx.x] = s;
  __syncthreads();
  s = 0.f;
#pragma unroll
  for (int w = 0; w < 8; ++w) s += red[w][threadIdx.x];
  const float inv = dead ? 0.f : 1.f / s;

  for (int q = qy; q < L; q += 8) {
    float val = dead ? 0.f : __expf(base[(size_t)q * L + k] - mx) * inv;
    outb[(size_t)q * L + k] = (_Float16)val;
  }
}

// ---- Launcher ---------------------------------------------------------------
extern "C" void kernel_launch(void* const* d_in, const int* in_sizes, int n_in,
                              void* d_out, int out_size, void* d_ws, size_t ws_size,
                              hipStream_t stream) {
  constexpr int BATCH = 8, L = 2048, D = 128;
  constexpr size_t NLD = (size_t)BATCH * L * D;  // 2,097,152
  constexpr size_t NLL = (size_t)BATCH * L * L;  // 33,554,432
  const float SCALE = 0.08838834764831845f;      // 1/sqrt(128)

  const float* Q = (const float*)d_in[0];
  const float* Km = (const float*)d_in[1];
  const float* V = (const float*)d_in[2];
  const unsigned char* qmask = (const unsigned char*)d_in[3];
  const unsigned char* kmask = (const unsigned char*)d_in[4];

  float* out = (float*)d_out;
  float* attn = out;            // [B,Lq,D]
  float* aw = attn + NLD;       // [B,Lq,Lk]
  float* co_attn = aw + NLL;    // [B,Lq,D]
  float* co_w = co_attn + NLD;  // [B,Lq,Lq]

  char* wsb = (char*)d_ws;
  float* ws_logit = (float*)wsb;                               // 128 MB f32
  _Float16* ws_kq = (_Float16*)(wsb + NLL * 4);                // 64 MB
  _Float16* ws_aw = (_Float16*)(wsb + NLL * 4 + NLL * 2);      // 64 MB
  _Float16* ws_cw = (_Float16*)(wsb + NLL * 4 + 2 * NLL * 2);  // 64 MB
  _Float16* ws_qh = (_Float16*)(wsb + NLL * 4 + 3 * NLL * 2);  // 4 MB each
  _Float16* ws_kh = ws_qh + NLD;
  _Float16* ws_vh = ws_kh + NLD;

  const dim3 blk(128);

  // 0) pack Q,K,V to f16
  cvt_f16<<<dim3((NLD / 4 + 255) / 256), dim3(256), 0, stream>>>(Q, ws_qh, (int)(NLD / 4));
  cvt_f16<<<dim3((NLD / 4 + 255) / 256), dim3(256), 0, stream>>>(Km, ws_kh, (int)(NLD / 4));
  cvt_f16<<<dim3((NLD / 4 + 255) / 256), dim3(256), 0, stream>>>(V, ws_vh, (int)(NLD / 4));

  // 1) logit = scale*Q.K^T, mask -> -inf   (NT, TDM-fed)
  gemm_nt_tdm<true><<<dim3((L / 64) * (L / 64), BATCH), blk, SMEM_BYTES, stream>>>(
      ws_qh, ws_kh, ws_logit, (_Float16*)nullptr, L, L, D, SCALE, qmask, kmask);

  // 2) attn_weight = softmax_k(logit): f32 -> d_out (NT), f16 mirror -> ws
  softmax_rows<<<dim3(BATCH * L), dim3(256), 0, stream>>>(ws_logit, aw, ws_aw, L);

  // 3) kq_weight = softmax_q(logit) -> f16
  softmax_cols<<<dim3(L / 32, BATCH), dim3(32, 8), 0, stream>>>(ws_logit, ws_kq, L);

  // 4) attn = attn_weight @ V   (NN)
  gemm_nn_tdm<<<dim3((L / 64) * (D / 64), BATCH), blk, SMEM_BYTES, stream>>>(
      ws_aw, ws_vh, attn, L, D, L);

  // 5) co_weight = attn_weight @ kq_weight^T  (NT; f16 mirror for step 6)
  gemm_nt_tdm<false><<<dim3((L / 64) * (L / 64), BATCH), blk, SMEM_BYTES, stream>>>(
      ws_aw, ws_kq, co_w, ws_cw, L, L, L, 1.0f, nullptr, nullptr);

  // 6) co_attn = co_weight @ Q   (NN)
  gemm_nn_tdm<<<dim3((L / 64) * (D / 64), BATCH), blk, SMEM_BYTES, stream>>>(
      ws_cw, ws_qh, co_attn, L, D, L);
}
